// S4Core_46076409151774
// MI455X (gfx1250) — matrix operand order repaired and checked by reading, not compiled, not run on hardware
//
#include <hip/hip_runtime.h>
#include <hip/hip_bf16.h>

typedef __attribute__((ext_vector_type(16))) _Float16 v16h;
typedef __attribute__((ext_vector_type(8)))  float    v8f;
typedef __attribute__((ext_vector_type(4)))  unsigned int v4u;
typedef __attribute__((ext_vector_type(8)))  int      v8i;
typedef __attribute__((ext_vector_type(4)))  int      v4i;

#define DM     1024
#define LSEQ   4096
#define NB     4
#define NCHUNK 256      // LSEQ/16
#define CT2    32       // chunks per block (2 column tiles of 16)
#define NTILE2 8        // NCHUNK/CT2

// workspace layout (floats):
#define WS_TAPS 0               // [DM][32]
#define WS_E    32768           // [DM][16][2]
#define WS_P    65536           // [DM][16][2]
#define WS_A16  98304           // [DM][2]
#define WS_S    100352          // [NB][NCHUNK][DM][2]  (8 MiB; scanned in place -> X)

// ---------------- kernel 1: per-channel coefficient prep ----------------
__global__ void s4_prep(const float* __restrict__ Ar, const float* __restrict__ Ai,
                        const float* __restrict__ Br, const float* __restrict__ Bi,
                        const float* __restrict__ Cr, const float* __restrict__ Ci,
                        float* __restrict__ ws) {
  int d = blockIdx.x * blockDim.x + threadIdx.x;
  if (d >= DM) return;
  float ar = Ar[d], ai = Ai[d];
  float br = Br[d], bi = Bi[d];
  float cr = Cr[d], ci = Ci[d];
  // w = C*B ; taps[t] = Re(w * a^t), t = 0..31
  float sr = cr * br - ci * bi, si = cr * bi + ci * br;
  float* taps = ws + WS_TAPS + d * 32;
  taps[0] = sr;
  for (int t = 1; t < 32; ++t) {
    float nr = sr * ar - si * ai, ni = sr * ai + si * ar;
    sr = nr; si = ni; taps[t] = sr;
  }
  // E[j] = a^(15-j) * B
  float* E = ws + WS_E + d * 32;
  float pr = br, pi = bi;
  E[15 * 2] = pr; E[15 * 2 + 1] = pi;
  for (int j = 14; j >= 0; --j) {
    float nr = pr * ar - pi * ai, ni = pr * ai + pi * ar;
    pr = nr; pi = ni; E[j * 2] = pr; E[j * 2 + 1] = pi;
  }
  // a^16
  float qr = 1.f, qi = 0.f;
  for (int t = 0; t < 16; ++t) {
    float nr = qr * ar - qi * ai, ni = qr * ai + qi * ar;
    qr = nr; qi = ni;
  }
  ws[WS_A16 + d * 2] = qr; ws[WS_A16 + d * 2 + 1] = qi;
  // P[i] = C * a^(i+17)
  float t17r = qr * ar - qi * ai, t17i = qr * ai + qi * ar;
  float gr = cr * t17r - ci * t17i, gi = cr * t17i + ci * t17r;
  float* P = ws + WS_P + d * 32;
  P[0] = gr; P[1] = gi;
  for (int i = 1; i < 16; ++i) {
    float nr = gr * ar - gi * ai, ni = gr * ai + gi * ar;
    gr = nr; gi = ni; P[i * 2] = gr; P[i * 2 + 1] = gi;
  }
}

// ---------------- kernel 2: per-chunk partial states S_e ----------------
__global__ void s4_state_partial(const float* __restrict__ u, float* __restrict__ ws) {
  int d = blockIdx.x * 256 + threadIdx.x;
  int e = blockIdx.y;
  int b = blockIdx.z;
  const float* E = ws + WS_E + d * 32;
  const float* up = u + ((size_t)b * LSEQ + (size_t)e * 16) * DM + d;
  float sr = 0.f, si = 0.f;
#pragma unroll
  for (int j = 0; j < 16; ++j) {
    float uu = up[(size_t)j * DM];       // coalesced over d
    sr += E[j * 2] * uu;
    si += E[j * 2 + 1] * uu;
  }
  size_t o = WS_S + (((size_t)b * NCHUNK + e) * DM + d) * 2;
  ws[o] = sr; ws[o + 1] = si;
}

// ---------------- kernel 3: in-place state scan X_e = a16*X_{e-1} + S_e ----------------
__global__ void s4_state_scan(float* __restrict__ ws) {
  int d = blockIdx.x * 256 + threadIdx.x;
  int b = blockIdx.y;
  float a16r = ws[WS_A16 + d * 2], a16i = ws[WS_A16 + d * 2 + 1];
  float xr = 0.f, xi = 0.f;
  float* Sb = ws + WS_S + ((size_t)b * NCHUNK) * DM * 2 + (size_t)d * 2;
  for (int e0 = 0; e0 < NCHUNK; e0 += 16) {
    float lr[16], li[16];
#pragma unroll
    for (int k = 0; k < 16; ++k) {       // batch loads (latency hiding)
      lr[k] = Sb[(size_t)(e0 + k) * DM * 2];
      li[k] = Sb[(size_t)(e0 + k) * DM * 2 + 1];
    }
#pragma unroll
    for (int k = 0; k < 16; ++k) {       // dependent chain is cheap VALU
      float nr = a16r * xr - a16i * xi + lr[k];
      float ni = a16r * xi + a16i * xr + li[k];
      xr = nr; xi = ni; lr[k] = xr; li[k] = xi;
    }
#pragma unroll
    for (int k = 0; k < 16; ++k) {
      Sb[(size_t)(e0 + k) * DM * 2]     = lr[k];
      Sb[(size_t)(e0 + k) * DM * 2 + 1] = li[k];
    }
  }
}

// ---------------- kernel 4: TDM staging + WMMA Toeplitz + inter-chunk + D*u ----------------
__global__ void __launch_bounds__(512) s4_conv_wmma(const float* __restrict__ u,
                                                    const float* __restrict__ Dp,
                                                    const float* __restrict__ ws,
                                                    float* __restrict__ y) {
  __shared__ float lu[528 * 17];    // u tile: 33 chunks x 16 l x 16 d (+1 pad per row)
  __shared__ float lyt[512 * 17];   // output transpose buffer

  int tid  = threadIdx.x;
  int w    = tid >> 5;              // wave id 0..15 -> channel within group
  int lane = tid & 31;
  int d0   = blockIdx.x * 16;
  int d    = d0 + w;
  int b    = blockIdx.y >> 3;       // NTILE2 = 8
  int ct   = blockIdx.y & 7;
  int c0   = ct * CT2;              // first chunk of the 32-chunk block
  int lbase = c0 * 16 - 16;         // includes previous chunk

  // --- zero-fill leading 16 rows when there is no previous chunk ---
  if (ct == 0) {
    for (int idx = tid; idx < 256; idx += 512) {
      lu[(idx >> 4) * 17 + (idx & 15)] = 0.f;
    }
  }

  // --- TDM: DMA the u tile (nrows x 16 f32, row stride DM) into LDS with
  //     1-DWORD padding every 16 DWORDs (pad_interval=3, pad_amount=0) ---
  if (tid < 32) {                   // one wave issues the descriptor
    int zrow   = (ct == 0) ? 16 : 0;
    int nrows  = 528 - zrow;
    int gstart = lbase + zrow;      // == max(lbase, 0)
    unsigned long long gaddr =
        (unsigned long long)(const void*)(u + ((size_t)b * LSEQ + gstart) * DM + d0);
    unsigned ldsoff = (unsigned)(size_t)(const void*)lu + (unsigned)zrow * 17u * 4u;
    // D# group 0: count=1 | lds_addr | global_addr[56:0] | type=2
    v4u g0 = { 1u, ldsoff,
               (unsigned)(gaddr & 0xffffffffu),
               (unsigned)((gaddr >> 32) & 0x1ffffffu) | (2u << 30) };
    // D# group 1: data_size=4B, pad_enable, pad_interval=16 DW, pad_amount=1 DW,
    //             tensor_dim0=16, tensor_dim1=nrows, tile_dim0=16, tile_dim1=nrows,
    //             tensor_dim0_stride=1024
    v8i g1 = { (int)0x00D20000,          // mask=0 | data_size=2<<16 | pad en/intv/amt
               (int)(16 << 16),          // tensor_dim0[15:0] at bit 48
               (int)(nrows << 16),       // tensor_dim1[15:0] at bit 80
               (int)(16 << 16),          // tile_dim0 at bits 127:112
               nrows,                    // tile_dim1 at bits 143:128
               1024,                     // tensor_dim0_stride[31:0]
               0, 0 };
    v4i gz4 = { 0, 0, 0, 0 };
    v8i gz8 = { 0, 0, 0, 0, 0, 0, 0, 0 };
    __builtin_amdgcn_tensor_load_to_lds(g0, g1, gz4, gz4, gz8, 0);
    __builtin_amdgcn_s_wait_tensorcnt(0);
  }
  __syncthreads();

  int g = lane >> 4;                // half-wave group
  int m = lane & 15;                // output row for A; column index for B
  int n = lane & 15;

  // --- A operand: [M2 | M1]_d, A[m][k] = (16+m-k >= 0) ? taps[16+m-k] : 0 ---
  const float* taps = ws + WS_TAPS + d * 32;
  v16h amat;
#pragma unroll
  for (int e = 0; e < 16; ++e) {
    int k = 8 * g + e + ((e < 8) ? 0 : 8);     // documented 16-bit A layout
    int T = 16 + m - k;                        // T <= 31 always here
    int Tc = (T < 0) ? 0 : T;
    float v = taps[Tc];
    v = (T < 0) ? 0.f : v;
    amat[e] = (_Float16)v;
  }

  float dd = Dp[d];
  const float* P = ws + WS_P + d * 32;

#pragma unroll
  for (int t2 = 0; t2 < 2; ++t2) {
    // --- B operand: 32x16, k-rows = [u_{c-1} ; u_c], col n -> chunk c0+16*t2+n ---
    v16h bmat;
#pragma unroll
    for (int e = 0; e < 16; ++e) {
      int k = 8 * g + e + ((e < 8) ? 0 : 8);   // documented B layout
      bmat[e] = (_Float16)lu[(256 * t2 + 16 * n + k) * 17 + w];
    }

    v8f acc = {};
    acc = __builtin_amdgcn_wmma_f32_16x16x32_f16(
        /*neg_a=*/false, amat, /*neg_b=*/false, bmat,
        /*c_mod=*/(short)0, acc, /*reuse_a=*/false, /*reuse_b=*/false);

    // --- long-range term: Re(P[m] * X_{c-2}) ; plus D*u ---
    int c = c0 + 16 * t2 + n;
    float xr = 0.f, xi = 0.f;
    if (c >= 2) {
      size_t o = WS_S + (((size_t)b * NCHUNK + (c - 2)) * DM + d) * 2;
      xr = ws[o]; xi = ws[o + 1];
    }
#pragma unroll
    for (int r = 0; r < 8; ++r) {
      int mr = r + 8 * g;                      // documented f32 C/D layout
      float v = acc[r];
      v += P[mr * 2] * xr - P[mr * 2 + 1] * xi;
      v += dd * lu[(256 * t2 + 16 * n + 16 + mr) * 17 + w];
      lyt[(t2 * 256 + n * 16 + mr) * 17 + w] = v;
    }
  }
  __syncthreads();

  // --- coalesced store: contiguous 16 d's per (l) row ---
  for (int it = 0; it < 16; ++it) {
    int idx = it * 512 + tid;                  // 8192 outputs per block
    int row = idx >> 4, wc = idx & 15;
    int n2 = row >> 4, m2 = row & 15;
    int l = (c0 + n2) * 16 + m2;
    y[((size_t)b * LSEQ + l) * DM + d0 + wc] = lyt[row * 17 + wc];
  }
}

// ---------------- launcher ----------------
extern "C" void kernel_launch(void* const* d_in, const int* in_sizes, int n_in,
                              void* d_out, int out_size, void* d_ws, size_t ws_size,
                              hipStream_t stream) {
  const float* u  = (const float*)d_in[0];
  const float* Ar = (const float*)d_in[1];
  const float* Ai = (const float*)d_in[2];
  const float* Br = (const float*)d_in[3];
  const float* Bi = (const float*)d_in[4];
  const float* Cr = (const float*)d_in[5];
  const float* Ci = (const float*)d_in[6];
  const float* Dp = (const float*)d_in[7];
  float* y   = (float*)d_out;
  float* wsf = (float*)d_ws;

  s4_prep<<<dim3(DM / 256), dim3(256), 0, stream>>>(Ar, Ai, Br, Bi, Cr, Ci, wsf);
  s4_state_partial<<<dim3(DM / 256, NCHUNK, NB), dim3(256), 0, stream>>>(u, wsf);
  s4_state_scan<<<dim3(DM / 256, NB), dim3(256), 0, stream>>>(wsf);
  s4_conv_wmma<<<dim3(DM / 16, NB * NTILE2), dim3(512), 0, stream>>>(u, Dp, wsf, y);
}
